// GCN_44650480009986
// MI455X (gfx1250) — matrix operand order repaired and checked by reading, not compiled, not run on hardware
//
#include <hip/hip_runtime.h>
#include <hip/hip_bf16.h>

typedef __attribute__((ext_vector_type(4)))  _Float16 v4h;
typedef __attribute__((ext_vector_type(8)))  _Float16 v8h;
typedef __attribute__((ext_vector_type(16))) _Float16 v16h;
typedef __attribute__((ext_vector_type(8)))  float    v8f;

#define N_NODES 160000
#define N_EDGES 2560000
#define ETOT    (N_EDGES + N_NODES)   // edges + self loops
#define NB      8
#define HID     512
#define OUTF    256
#define NL      12

// ---------------------------------------------------------------- utilities
__global__ void k_fill_u32(unsigned* __restrict__ p, unsigned v, int n) {
    int i = blockIdx.x * 256 + threadIdx.x;
    if (i < n) p[i] = v;
}

// ------------------------------------------------- degree (incl. self loops)
__global__ void k_deg(const int* __restrict__ ei, unsigned* __restrict__ deg) {
    int e = blockIdx.x * 256 + threadIdx.x;
    if (e >= ETOT) return;
    int dst = (e < N_EDGES) ? ei[N_EDGES + e] : (e - N_EDGES);
    atomicAdd(&deg[dst], 1u);
}

__global__ void k_dinv(const unsigned* __restrict__ deg, float* __restrict__ dinv) {
    int i = blockIdx.x * 256 + threadIdx.x;
    if (i >= N_NODES) return;
    unsigned d = deg[i];
    dinv[i] = (d > 0u) ? rsqrtf((float)d) : 0.0f;
}

// --------------------------------------- single-block scan: rowptr from deg
__global__ void __launch_bounds__(1024)
k_scan(const unsigned* __restrict__ deg, unsigned* __restrict__ rowptr, int n) {
    __shared__ unsigned sh[1024];
    __shared__ unsigned carry;
    if (threadIdx.x == 0) { carry = 0u; rowptr[0] = 0u; }
    __syncthreads();
    for (int base = 0; base < n; base += 1024) {
        int i = base + (int)threadIdx.x;
        sh[threadIdx.x] = (i < n) ? deg[i] : 0u;
        __syncthreads();
        for (int off = 1; off < 1024; off <<= 1) {
            unsigned t = (threadIdx.x >= (unsigned)off) ? sh[threadIdx.x - off] : 0u;
            __syncthreads();
            sh[threadIdx.x] += t;
            __syncthreads();
        }
        if (i < n) rowptr[i + 1] = carry + sh[threadIdx.x];
        __syncthreads();
        if (threadIdx.x == 1023) carry += sh[1023];
        __syncthreads();
    }
}

// -------------------------------------------- bin edges into dst-major CSR
__global__ void k_bin(const int* __restrict__ ei, const float* __restrict__ dinv,
                      const unsigned* __restrict__ rowptr, unsigned* __restrict__ cursor,
                      unsigned* __restrict__ csr_src, float* __restrict__ csr_w) {
    int e = blockIdx.x * 256 + threadIdx.x;
    if (e >= ETOT) return;
    int src, dst;
    if (e < N_EDGES) { src = ei[e]; dst = ei[N_EDGES + e]; }
    else             { src = dst = e - N_EDGES; }
    float wv = dinv[src] * dinv[dst];
    unsigned pos  = atomicAdd(&cursor[dst], 1u);
    unsigned slot = rowptr[dst] + pos;
    csr_src[slot] = (unsigned)src;
    csr_w[slot]   = wv;
}

// ----------------- layer-1 scalar aggregation: s[d] = sum_e w_e * x[src_e]
__global__ void k_saggr(const float* __restrict__ x, const unsigned* __restrict__ rowptr,
                        const unsigned* __restrict__ csr_src, const float* __restrict__ csr_w,
                        float* __restrict__ s) {
    int d = blockIdx.x * 256 + threadIdx.x;
    if (d >= N_NODES) return;
    unsigned b = rowptr[d], e = rowptr[d + 1];
    float acc = 0.0f;
    for (unsigned k = b; k < e; ++k) acc += csr_w[k] * x[csr_src[k]];
    s[d] = acc;
}

// ------------------- h1[d,j] = relu(s[d]*w1[j] + b1[j]) stored as f16 rows
__global__ void k_h1(const float* __restrict__ s, const float* __restrict__ w1,
                     const float* __restrict__ b1, _Float16* __restrict__ h) {
    int idx = blockIdx.x * 256 + threadIdx.x;          // < N*HID
    int d = idx >> 9, j = idx & (HID - 1);
    float v = fmaf(s[d], w1[j], b1[j]);
    h[(size_t)idx] = (_Float16)fmaxf(v, 0.0f);
}

// --------- pack w2 (f32 [K=512][N=512] row-major) into WMMA B-frag layout:
// tile (kt,ct): lane l holds B[kt*32 + (i<8?0:16) + (l<16?0:8) + (i&7)][ct*16 + (l&15)], i=0..15
__global__ void k_packW(const float* __restrict__ w2, _Float16* __restrict__ wp) {
    int idx = blockIdx.x * 256 + threadIdx.x;          // < 512*512
    int i    = idx & 15;
    int lane = (idx >> 4) & 31;
    int t    = idx >> 9;
    int ct   = t & 31;
    int kt   = t >> 5;
    int K = kt * 32 + ((i < 8) ? 0 : 16) + ((lane < 16) ? 0 : 8) + (i & 7);
    int n = ct * 16 + (lane & 15);
    wp[(size_t)idx] = (_Float16)w2[(size_t)K * HID + n];
}

// ------------- WMMA GEMM: D[N,512] = A[N,512] @ W, f32 accum.
// 2x2 register blocking: each wave32 owns a 32x32 output block (4 WMMA tiles),
// so each A/B fragment load feeds two v_wmma ops (2 b128 loads per WMMA).
__global__ void __launch_bounds__(256)
k_gemm(const _Float16* __restrict__ A, const _Float16* __restrict__ Bp,
       _Float16* __restrict__ D) {
    int wave = (blockIdx.x * 256 + (int)threadIdx.x) >> 5;
    int lane = threadIdx.x & 31;
    int ct2 = wave & 15;          // 16 super-tiles of 32 columns
    int rt2 = wave >> 4;          // 5000 super-tiles of 32 rows
    int row0 = rt2 * 32, col0 = ct2 * 32;
    int lo  = lane & 15;
    bool low = lane < 16;

    const _Float16* ar0 = A + (size_t)(row0 + lo) * HID + (low ? 0 : 8);
    const _Float16* ar1 = ar0 + (size_t)16 * HID;
    // B fragment for tile (kt, ct): halves offset = (kt*32 + ct)*512 + lane*16
    const _Float16* bp0 = Bp + (size_t)(ct2 * 2) * 512 + (size_t)lane * 16;

    v8f acc00 = {}, acc01 = {}, acc10 = {}, acc11 = {};
#pragma unroll 2
    for (int kt = 0; kt < 16; ++kt) {
        v8h a0l = *(const v8h*)(ar0 + kt * 32);
        v8h a0h = *(const v8h*)(ar0 + kt * 32 + 16);
        v8h a1l = *(const v8h*)(ar1 + kt * 32);
        v8h a1h = *(const v8h*)(ar1 + kt * 32 + 16);
        v16h a0, a1;
#pragma unroll
        for (int i = 0; i < 8; ++i) {
            a0[i] = a0l[i]; a0[8 + i] = a0h[i];
            a1[i] = a1l[i]; a1[8 + i] = a1h[i];
        }
        const _Float16* bkt = bp0 + (size_t)kt * 32 * 512;
        v16h b0 = *(const v16h*)(bkt);
        v16h b1 = *(const v16h*)(bkt + 512);
        acc00 = __builtin_amdgcn_wmma_f32_16x16x32_f16(false, a0, false, b0,
                                                       (short)0, acc00, false, false);
        acc01 = __builtin_amdgcn_wmma_f32_16x16x32_f16(false, a0, false, b1,
                                                       (short)0, acc01, false, false);
        acc10 = __builtin_amdgcn_wmma_f32_16x16x32_f16(false, a1, false, b0,
                                                       (short)0, acc10, false, false);
        acc11 = __builtin_amdgcn_wmma_f32_16x16x32_f16(false, a1, false, b1,
                                                       (short)0, acc11, false, false);
    }
    int mbase = low ? 0 : 8;
#pragma unroll
    for (int r = 0; r < 8; ++r) {
        size_t ro  = (size_t)(row0 + mbase + r) * HID;
        size_t ro1 = ro + (size_t)16 * HID;
        D[ro  + col0 + lo]      = (_Float16)acc00[r];
        D[ro  + col0 + 16 + lo] = (_Float16)acc01[r];
        D[ro1 + col0 + lo]      = (_Float16)acc10[r];
        D[ro1 + col0 + 16 + lo] = (_Float16)acc11[r];
    }
}

// ------- CSR gather-aggregation (no wide atomics): out[d,:] = sum w*hw[src,:]
__global__ void __launch_bounds__(128)
k_prop(const _Float16* __restrict__ hw, const unsigned* __restrict__ rowptr,
       const unsigned* __restrict__ csr_src, const float* __restrict__ csr_w,
       const float* __restrict__ bias, _Float16* __restrict__ out, int do_relu) {
    int d = blockIdx.x;
    int f = (int)threadIdx.x * 4;                  // 128 threads * 4 feats = 512
    unsigned b = rowptr[d], e = rowptr[d + 1];
    float a0 = 0.f, a1 = 0.f, a2 = 0.f, a3 = 0.f;
    for (unsigned k = b; k < e; ++k) {
        unsigned sn = csr_src[k];
        float w = csr_w[k];
        v4h p = *(const v4h*)(hw + (size_t)sn * HID + f);
        a0 = fmaf(w, (float)p[0], a0);
        a1 = fmaf(w, (float)p[1], a1);
        a2 = fmaf(w, (float)p[2], a2);
        a3 = fmaf(w, (float)p[3], a3);
    }
    a0 += bias[f]; a1 += bias[f + 1]; a2 += bias[f + 2]; a3 += bias[f + 3];
    if (do_relu) {
        a0 = fmaxf(a0, 0.f); a1 = fmaxf(a1, 0.f);
        a2 = fmaxf(a2, 0.f); a3 = fmaxf(a3, 0.f);
    }
    v4h o; o[0] = (_Float16)a0; o[1] = (_Float16)a1; o[2] = (_Float16)a2; o[3] = (_Float16)a3;
    *(v4h*)(out + (size_t)d * HID + f) = o;
}

// ------------------------------------------------------- per-graph counts
__global__ void k_cnt(const int* __restrict__ batch, unsigned* __restrict__ cnt) {
    __shared__ unsigned lc[NB];
    if (threadIdx.x < NB) lc[threadIdx.x] = 0u;
    __syncthreads();
    int i = blockIdx.x * 256 + threadIdx.x;
    if (i < N_NODES) atomicAdd(&lc[batch[i]], 1u);
    __syncthreads();
    if (threadIdx.x < NB && lc[threadIdx.x]) atomicAdd(&cnt[threadIdx.x], lc[threadIdx.x]);
}

// ------------- pooled sums: register accumulators per graph, few atomics
__global__ void __launch_bounds__(256)
k_pool(const _Float16* __restrict__ h3, const int* __restrict__ batch,
       float* __restrict__ pooled) {
    int j  = (blockIdx.x & 1) * 256 + (int)threadIdx.x;   // feature
    int c  = blockIdx.x >> 1;                             // 64 chunks of 2500 nodes
    int n0 = c * 2500, n1 = n0 + 2500;
    float acc[NB];
#pragma unroll
    for (int g = 0; g < NB; ++g) acc[g] = 0.0f;
    for (int n = n0; n < n1; ++n)
        acc[batch[n]] += (float)h3[(size_t)n * HID + j];
#pragma unroll
    for (int g = 0; g < NB; ++g) atomicAdd(&pooled[g * HID + j], acc[g]);
}

// ------------------------- FiLM heads: out[2,L,B,OUT] = mean @ W{g,b} + b
__global__ void k_head(const float* __restrict__ pooled, const unsigned* __restrict__ cnt,
                       const float* __restrict__ Wg, const float* __restrict__ bg,
                       const float* __restrict__ Wb, const float* __restrict__ bb,
                       float* __restrict__ out) {
    int idx = blockIdx.x * 256 + threadIdx.x;             // < 2*12*8*256
    if (idx >= 2 * NL * NB * OUTF) return;
    int o  = idx & (OUTF - 1);
    int g  = (idx >> 8) & (NB - 1);
    int ls = idx >> 11;
    int l  = ls % NL;
    int sel = ls / NL;
    const float* W = sel ? Wb : Wg;
    const float* B = sel ? bb : bg;
    float invc = 1.0f / fmaxf((float)cnt[g], 1.0f);
    const float* pg = pooled + g * HID;
    const float* Wl = W + (size_t)l * HID * OUTF + o;
    float acc = 0.0f;
    for (int h = 0; h < HID; ++h)
        acc = fmaf(pg[h] * invc, Wl[(size_t)h * OUTF], acc);
    out[idx] = acc + B[l * OUTF + o];
}

// ---------------------------------------------------------------- launcher
extern "C" void kernel_launch(void* const* d_in, const int* in_sizes, int n_in,
                              void* d_out, int out_size, void* d_ws, size_t ws_size,
                              hipStream_t stream) {
    const float* x     = (const float*)d_in[0];
    const int*   ei    = (const int*)  d_in[1];
    const int*   batch = (const int*)  d_in[2];
    const float* w1    = (const float*)d_in[3];
    const float* b1    = (const float*)d_in[4];
    const float* w2    = (const float*)d_in[5];
    const float* b2    = (const float*)d_in[6];
    const float* Wg    = (const float*)d_in[7];
    const float* bg    = (const float*)d_in[8];
    const float* Wb    = (const float*)d_in[9];
    const float* bb    = (const float*)d_in[10];
    float* out = (float*)d_out;
    (void)in_sizes; (void)n_in; (void)out_size; (void)ws_size;

    char* base = (char*)d_ws;
    size_t off = 0;
    auto alloc = [&](size_t bytes) -> void* {
        void* p = base + off;
        off += (bytes + 255) & ~(size_t)255;
        return p;
    };
    unsigned* deg     = (unsigned*)alloc((size_t)N_NODES * 4);
    unsigned* cursor  = (unsigned*)alloc((size_t)N_NODES * 4);
    unsigned* rowptr  = (unsigned*)alloc((size_t)(N_NODES + 1) * 4);
    float*    dinv    = (float*)   alloc((size_t)N_NODES * 4);
    float*    sagg    = (float*)   alloc((size_t)N_NODES * 4);
    unsigned* csr_src = (unsigned*)alloc((size_t)ETOT * 4);
    float*    csr_w   = (float*)   alloc((size_t)ETOT * 4);
    _Float16* wpack   = (_Float16*)alloc((size_t)HID * HID * 2);
    unsigned* cnt     = (unsigned*)alloc(NB * 4);
    float*    pooled  = (float*)   alloc(NB * HID * 4);
    _Float16* hbuf    = (_Float16*)alloc((size_t)N_NODES * HID * 2);  // h1 / h2 / h3
    _Float16* hwbuf   = (_Float16*)alloc((size_t)N_NODES * HID * 2);  // h@W (gather src)

    const int TPB = 256;
    // zero accumulators / cursors (ws is not re-poisoned between replays)
    k_fill_u32<<<(N_NODES + TPB - 1) / TPB, TPB, 0, stream>>>(deg, 0u, N_NODES);
    k_fill_u32<<<(N_NODES + TPB - 1) / TPB, TPB, 0, stream>>>(cursor, 0u, N_NODES);
    k_fill_u32<<<1, TPB, 0, stream>>>(cnt, 0u, NB);
    k_fill_u32<<<(NB * HID + TPB - 1) / TPB, TPB, 0, stream>>>((unsigned*)pooled, 0u, NB * HID);

    // GCN normalization + CSR build (reused by all three convs)
    k_deg <<<(ETOT + TPB - 1) / TPB, TPB, 0, stream>>>(ei, deg);
    k_dinv<<<(N_NODES + TPB - 1) / TPB, TPB, 0, stream>>>(deg, dinv);
    k_scan<<<1, 1024, 0, stream>>>(deg, rowptr, N_NODES);
    k_bin <<<(ETOT + TPB - 1) / TPB, TPB, 0, stream>>>(ei, dinv, rowptr, cursor, csr_src, csr_w);

    // layer 1 collapsed to scalar aggregation + outer product
    k_saggr<<<(N_NODES + TPB - 1) / TPB, TPB, 0, stream>>>(x, rowptr, csr_src, csr_w, sagg);
    k_h1<<<(N_NODES * HID) / TPB, TPB, 0, stream>>>(sagg, w1, b1, hbuf);

    // pack w2 once into WMMA B-fragment layout (reused by conv2 and conv3)
    k_packW<<<(HID * HID) / TPB, TPB, 0, stream>>>(w2, wpack);

    const int gemm_blocks = (N_NODES / 32) * (HID / 32) / 8;   // 8 waves/block, 32x32/wave
    // conv2: GEMM + propagate (+b2, relu) -> h2
    k_gemm<<<gemm_blocks, TPB, 0, stream>>>(hbuf, wpack, hwbuf);
    k_prop<<<N_NODES, 128, 0, stream>>>(hwbuf, rowptr, csr_src, csr_w, b2, hbuf, 1);
    // conv3: GEMM + propagate (+b2, no relu) -> h3
    k_gemm<<<gemm_blocks, TPB, 0, stream>>>(hbuf, wpack, hwbuf);
    k_prop<<<N_NODES, 128, 0, stream>>>(hwbuf, rowptr, csr_src, csr_w, b2, hbuf, 0);

    // global mean pool + FiLM heads
    k_cnt <<<(N_NODES + TPB - 1) / TPB, TPB, 0, stream>>>(batch, cnt);
    k_pool<<<128, TPB, 0, stream>>>(hbuf, batch, pooled);
    k_head<<<(2 * NL * NB * OUTF) / TPB, TPB, 0, stream>>>(pooled, cnt, Wg, bg, Wb, bb, out);
}
